// Model_66331474920155
// MI455X (gfx1250) — matrix operand order repaired and checked by reading, not compiled, not run on hardware
//
#include <hip/hip_runtime.h>
#include <hip/hip_bf16.h>

// ---------------------------------------------------------------------------
// Types / WMMA helpers (CDNA5 gfx1250, wave32)
// ---------------------------------------------------------------------------
typedef __attribute__((ext_vector_type(16))) __bf16 v16bf;
typedef __attribute__((ext_vector_type(8)))  __bf16 v8bf;
typedef __attribute__((ext_vector_type(8)))  float  v8f;

union Frag16 { v16bf v; v8bf h[2]; };
union PackBf2 { __bf16 h[2]; unsigned u; };

static __device__ inline v8f vzero8() {
    v8f z;
#pragma unroll
    for (int i = 0; i < 8; ++i) z[i] = 0.f;
    return z;
}

// Convert 8 consecutive fp32 (32B-aligned) to a v8bf
static __device__ inline v8bf cvt8(const float* g) {
    float4 a0 = *(const float4*)g;
    float4 a1 = *(const float4*)(g + 4);
    v8bf p;
    p[0] = (__bf16)a0.x; p[1] = (__bf16)a0.y; p[2] = (__bf16)a0.z; p[3] = (__bf16)a0.w;
    p[4] = (__bf16)a1.x; p[5] = (__bf16)a1.y; p[6] = (__bf16)a1.z; p[7] = (__bf16)a1.w;
    return p;
}

// A-matrix 16x32 bf16 fragment (ISA 7.12.2): lane m = lane&15;
// lanes 0-15 hold K {kb+0..7, kb+16..23}, lanes 16-31 hold K {kb+8..15, kb+24..31}.
static __device__ inline v16bf load_afrag(const __bf16* row, int kbase, int lane) {
    const int off = (lane >> 4) << 3;
    Frag16 u;
    u.h[0] = *(const v8bf*)(row + kbase + off);
    u.h[1] = *(const v8bf*)(row + kbase + 16 + off);
    return u.v;
}

// B-matrix 32x16 bf16 fragment: n = lane&15; lanes 0-15 K {kb+0..7, kb+16..23},
// lanes 16-31 K {kb+8..15, kb+24..31}.  base[n*stride + k] must hold B(k, n).
static __device__ inline v16bf load_bfrag(const __bf16* base, int stride, int kbase, int lane) {
    const __bf16* p = base + (lane & 15) * stride + kbase + ((lane >> 4) << 3);
    Frag16 u;
    u.h[0] = *(const v8bf*)(p);
    u.h[1] = *(const v8bf*)(p + 16);
    return u.v;
}

static __device__ inline v8f wmma_bf16(v16bf a, v16bf b, v8f c) {
    return __builtin_amdgcn_wmma_f32_16x16x32_bf16(false, a, false, b, (short)0, c, false, false);
}

// Async global->LDS copy of 16 bytes (gfx1250 GLOBAL_LOAD_ASYNC_TO_LDS_B128,
// tracked by ASYNCcnt).  ldsAddr = LDS byte address (low 32 bits of the
// generic shared-aperture pointer per ISA 10.2 aperture mapping).
static __device__ inline void async_copy_b128(void* lds_dst, const void* gsrc) {
    unsigned laddr = (unsigned)(unsigned long long)lds_dst;
    asm volatile("global_load_async_to_lds_b128 %0, %1, off"
                 :: "v"(laddr), "v"(gsrc) : "memory");
}
static __device__ inline void wait_asynccnt0() {
    asm volatile("s_wait_asynccnt 0x0" ::: "memory");
}

// ---------------------------------------------------------------------------
// Model constants
// ---------------------------------------------------------------------------
#define BATCH   16
#define SEQ     1024
#define DMODEL  512
#define NHEAD   8
#define DHEAD   64
#define DFFN    2048
#define MROWS   (BATCH * SEQ)     // 16384
#define PREDLEN 512
#define COUT    21
#define CIN     21

// ---------------------------------------------------------------------------
// Weight pack: fp32 [K,N] -> bf16 [N,K] (transposed)
// ---------------------------------------------------------------------------
__launch_bounds__(256)
__global__ void pack_wt_kernel(const float* __restrict__ W, __bf16* __restrict__ Wt,
                               int K, int N) {
    int idx = blockIdx.x * 256 + threadIdx.x;
    if (idx >= K * N) return;
    int n = idx % N, k = idx / N;
    Wt[(size_t)n * K + k] = (__bf16)W[idx];
}

// ---------------------------------------------------------------------------
// Embedding: circular conv1d(k=3) + sinusoidal pos + temporal linear
// ---------------------------------------------------------------------------
__launch_bounds__(256)
__global__ void embed_kernel(const float* __restrict__ x, const float* __restrict__ mark,
                             const float* __restrict__ convw, const float* __restrict__ tempw,
                             float* __restrict__ out) {
    int idx = blockIdx.x * 256 + threadIdx.x;
    if (idx >= BATCH * SEQ * DMODEL) return;
    int d = idx & (DMODEL - 1);
    int bl = idx >> 9;
    int l = bl & (SEQ - 1);
    int b = bl >> 10;
    float acc = 0.f;
#pragma unroll
    for (int kw = 0; kw < 3; ++kw) {
        int src = (l + kw - 1 + SEQ) & (SEQ - 1);
        const float* xr = x + ((size_t)b * SEQ + src) * CIN;
        const float* wr = convw + (size_t)(kw * CIN) * DMODEL + d;
        for (int c = 0; c < CIN; ++c) acc += xr[c] * wr[(size_t)c * DMODEL];
    }
    float freq = __expf(-(float)((d >> 1) << 1) * (logf(10000.f) / (float)DMODEL));
    float ang = (float)l * freq;
    acc += (d & 1) ? __cosf(ang) : __sinf(ang);
    const float* mr = mark + ((size_t)b * SEQ + l) * 4;
#pragma unroll
    for (int t = 0; t < 4; ++t) acc += mr[t] * tempw[t * DMODEL + d];
    out[idx] = acc;
}

// ---------------------------------------------------------------------------
// Momentum bias: m[b,l] = mean_c(x[b,l,c]-x[b,l-3,c]), zeros for l<3
// ---------------------------------------------------------------------------
__launch_bounds__(256)
__global__ void momentum_kernel(const float* __restrict__ x, float* __restrict__ m) {
    int idx = blockIdx.x * 256 + threadIdx.x;
    if (idx >= BATCH * SEQ) return;
    int l = idx & (SEQ - 1);
    float r = 0.f;
    if (l >= 3) {
        const float* a = x + (size_t)idx * CIN;
        const float* p = x + (size_t)(idx - 3) * CIN;
        float s = 0.f;
        for (int c = 0; c < CIN; ++c) s += a[c] - p[c];
        r = s * (1.0f / (float)CIN);
    }
    m[idx] = r;
}

// ---------------------------------------------------------------------------
// bf16 WMMA GEMM: C[M,N] = act(A[M,K] @ W + bias),  W given transposed bf16 [N,K]
// Block 256 thr (8 waves), tile 128x128, K-step 32; wave tile 64x32 (8 accums).
// A staged fp32->bf16 in 8-elem vector chunks; W staged via async global->LDS.
// ---------------------------------------------------------------------------
#define ASTR 40   // LDS row stride in bf16 elements (80B, 16B multiple)
__launch_bounds__(256)
__global__ void gemm_bf16_kernel(const float* __restrict__ A, const __bf16* __restrict__ Wt,
                                 const float* __restrict__ bias, float* __restrict__ C,
                                 int M, int N, int K, int gelu) {
    __shared__ __align__(16) __bf16 ast[128 * ASTR];
    __shared__ __align__(16) __bf16 wst[128 * ASTR];
    const int tid = threadIdx.x, lane = tid & 31, wid = tid >> 5;
    const int m0 = blockIdx.x * 128, n0 = blockIdx.y * 128;
    const int wm = (wid & 1) * 64, wn = (wid >> 1) * 32;

    v8f acc[4][2];
#pragma unroll
    for (int mt = 0; mt < 4; ++mt) { acc[mt][0] = vzero8(); acc[mt][1] = vzero8(); }

    for (int k0 = 0; k0 < K; k0 += 32) {
        __syncthreads();
        // 4096 elements per tile = 512 chunks of 8; 2 chunks per thread
#pragma unroll
        for (int i = 0; i < 2; ++i) {
            int ch = tid + (i << 8);             // 0..511
            int r = ch >> 2, c = (ch & 3) << 3;  // row 0..127, col {0,8,16,24}
            *(v8bf*)&ast[r * ASTR + c] = cvt8(&A[(size_t)(m0 + r) * K + k0 + c]);
            async_copy_b128(&wst[r * ASTR + c], &Wt[(size_t)(n0 + r) * K + k0 + c]);
        }
        if (k0 + 32 < K) {   // prefetch next k-step (global_prefetch_b8)
            __builtin_prefetch(&A[(size_t)(m0 + (tid & 127)) * K + k0 + 32], 0, 0);
            __builtin_prefetch(&Wt[(size_t)(n0 + (tid & 127)) * K + k0 + 32], 0, 0);
        }
        wait_asynccnt0();
        __syncthreads();
        v16bf bfr[2];
#pragma unroll
        for (int nt = 0; nt < 2; ++nt)
            bfr[nt] = load_bfrag(wst + (wn + nt * 16) * ASTR, ASTR, 0, lane);
#pragma unroll
        for (int mt = 0; mt < 4; ++mt) {
            v16bf af = load_afrag(ast + (wm + mt * 16 + (lane & 15)) * ASTR, 0, lane);
            acc[mt][0] = wmma_bf16(af, bfr[0], acc[mt][0]);
            acc[mt][1] = wmma_bf16(af, bfr[1], acc[mt][1]);
        }
    }

    const int nlo = lane & 15, mhi = (lane >> 4) * 8;   // C layout: VGPR r -> m=r (+8 upper half)
#pragma unroll
    for (int mt = 0; mt < 4; ++mt)
#pragma unroll
        for (int nt = 0; nt < 2; ++nt)
#pragma unroll
            for (int r = 0; r < 8; ++r) {
                int row = m0 + wm + mt * 16 + r + mhi;
                int col = n0 + wn + nt * 16 + nlo;
                float v = acc[mt][nt][r] + bias[col];
                if (gelu) v = 0.5f * v * (1.f + erff(v * 0.70710678f));
                C[(size_t)row * N + col] = v;
            }
}

// ---------------------------------------------------------------------------
// Attention: one block per (b, h, 64-query tile).  Full score rows in LDS
// (64x1024 f32 = 256KB of the 320KB WGP LDS), softmax in place, then P@V.
// Both phases use v_wmma_f32_16x16x32_bf16.
// ---------------------------------------------------------------------------
#define KSTR 72
#define VSTR 40
#define ATTN_SMEM (64 * SEQ * 4 + 4096 + 64 * 64 * 2 + 64 * KSTR * 2 + 64 * VSTR * 2) // 288768

__launch_bounds__(128)
__global__ void attn_kernel(const float* __restrict__ Q, const float* __restrict__ Kx,
                            const float* __restrict__ Vx, const float* __restrict__ keybias,
                            float* __restrict__ O, int causal) {
    extern __shared__ char smem[];
    float* sc  = (float*)smem;                                   // [64][SEQ]
    float* bia = (float*)(smem + 64 * SEQ * 4);                  // [SEQ]
    __bf16* qst = (__bf16*)(smem + 64 * SEQ * 4 + 4096);         // [64][64]
    __bf16* kst = qst + 64 * 64;                                 // [64][KSTR]
    __bf16* vst = kst + 64 * KSTR;                               // [64][VSTR] (transposed V)

    const int tid = threadIdx.x, lane = tid & 31, wid = tid >> 5;
    const int qb = blockIdx.x * 64, h = blockIdx.y, b = blockIdx.z;
    const size_t bhoff = (size_t)b * SEQ * DMODEL + (size_t)h * DHEAD;

    for (int i = tid; i < SEQ; i += 128) bia[i] = keybias[b * SEQ + i];
    // stage Q: 64x64 fp32 -> bf16, 8-elem vector chunks (512 chunks / 128 thr)
#pragma unroll
    for (int i = 0; i < 4; ++i) {
        int ch = tid + (i << 7);
        int r = ch >> 3, c = (ch & 7) << 3;
        *(v8bf*)&qst[r * 64 + c] = cvt8(&Q[bhoff + (size_t)(qb + r) * DMODEL + c]);
    }
    __syncthreads();

    v16bf qf[2];
    {
        const __bf16* row = qst + (wid * 16 + (lane & 15)) * 64;
        qf[0] = load_afrag(row, 0, lane);
        qf[1] = load_afrag(row, 32, lane);
    }

    // ---- scores: S = (Q K^T) * rsqrt(dh) + keybias (+ causal mask) ----
    for (int s0 = 0; s0 < SEQ; s0 += 64) {
        __syncthreads();
#pragma unroll
        for (int i = 0; i < 4; ++i) {
            int ch = tid + (i << 7);
            int r = ch >> 3, c = (ch & 7) << 3;
            *(v8bf*)&kst[r * KSTR + c] = cvt8(&Kx[bhoff + (size_t)(s0 + r) * DMODEL + c]);
        }
        __syncthreads();
#pragma unroll
        for (int nt = 0; nt < 4; ++nt) {
            v8f a = vzero8();
            v16bf b0 = load_bfrag(kst + nt * 16 * KSTR, KSTR, 0, lane);
            v16bf b1 = load_bfrag(kst + nt * 16 * KSTR, KSTR, 32, lane);
            a = wmma_bf16(qf[0], b0, a);
            a = wmma_bf16(qf[1], b1, a);
            const int scol = s0 + nt * 16 + (lane & 15);
            const int mhi = (lane >> 4) * 8;
#pragma unroll
            for (int r = 0; r < 8; ++r) {
                int mrow = wid * 16 + r + mhi;
                float v = a[r] * 0.125f + bia[scol];
                if (causal && scol > qb + mrow) v = -1e9f;
                sc[mrow * SEQ + scol] = v;
            }
        }
    }
    __syncthreads();

    // ---- softmax per row; write P as bf16 pairs in place (ascending-safe:
    //      packed write to f32 slot i consumes f32 slots 2i, 2i+1 >= i) ----
    if (tid < 64) {
        float* row = sc + tid * SEQ;
        float mx = -3.4e38f;
        for (int i = 0; i < SEQ; ++i) mx = fmaxf(mx, row[i]);
        float sum = 0.f;
        for (int i = 0; i < SEQ; ++i) { float e = expf(row[i] - mx); row[i] = e; sum += e; }
        float inv = 1.f / sum;
        unsigned* prow = (unsigned*)row;
        for (int i = 0; i < SEQ / 2; ++i) {
            PackBf2 pk;
            pk.h[0] = (__bf16)(row[2 * i] * inv);
            pk.h[1] = (__bf16)(row[2 * i + 1] * inv);
            prow[i] = pk.u;
        }
    }
    __syncthreads();

    // ---- P @ V ----
    v8f acc[4];
#pragma unroll
    for (int nt = 0; nt < 4; ++nt) acc[nt] = vzero8();
    for (int s0 = 0; s0 < SEQ; s0 += 32) {
        // stage V transposed: vst[d][s], 8 s-values per thread -> one b128 store
#pragma unroll
        for (int i = 0; i < 2; ++i) {
            int ch = tid + (i << 7);                 // 0..255
            int d = ch & 63, sb = (ch >> 6) << 3;    // sb in {0,8,16,24}
            v8bf p;
#pragma unroll
            for (int j = 0; j < 8; ++j)
                p[j] = (__bf16)Vx[bhoff + (size_t)(s0 + sb + j) * DMODEL + d];
            *(v8bf*)&vst[d * VSTR + sb] = p;
        }
        __syncthreads();
        const __bf16* prow = (const __bf16*)(sc + (wid * 16 + (lane & 15)) * SEQ);
        v16bf pf = load_afrag(prow, s0, lane);
#pragma unroll
        for (int nt = 0; nt < 4; ++nt) {
            v16bf bf = load_bfrag(vst + nt * 16 * VSTR, VSTR, 0, lane);
            acc[nt] = wmma_bf16(pf, bf, acc[nt]);
        }
        __syncthreads();
    }

    const int nlo = lane & 15, mhi = (lane >> 4) * 8;
#pragma unroll
    for (int nt = 0; nt < 4; ++nt)
#pragma unroll
        for (int r = 0; r < 8; ++r) {
            int mrow = qb + wid * 16 + r + mhi;
            O[bhoff + (size_t)mrow * DMODEL + nt * 16 + nlo] = acc[nt][r];
        }
}

// ---------------------------------------------------------------------------
// Fused residual-add + LayerNorm, one wave (32 lanes) per 512-elem row
// ---------------------------------------------------------------------------
__launch_bounds__(256)
__global__ void add_ln_kernel(const float* __restrict__ x, const float* __restrict__ y,
                              const float* __restrict__ g, const float* __restrict__ be,
                              float* __restrict__ out, int rows) {
    const int lane = threadIdx.x & 31, wid = threadIdx.x >> 5;
    const int row = blockIdx.x * 8 + wid;
    if (row >= rows) return;
    const float* xr = x + (size_t)row * DMODEL;
    const float* yr = y ? y + (size_t)row * DMODEL : nullptr;
    float v[16];
    float s = 0.f;
#pragma unroll
    for (int i = 0; i < 16; ++i) {
        float t = xr[lane * 16 + i];
        if (yr) t += yr[lane * 16 + i];
        v[i] = t; s += t;
    }
#pragma unroll
    for (int o = 16; o; o >>= 1) s += __shfl_xor(s, o, 32);
    float mu = s * (1.f / (float)DMODEL);
    float var = 0.f;
#pragma unroll
    for (int i = 0; i < 16; ++i) { float d0 = v[i] - mu; var += d0 * d0; }
#pragma unroll
    for (int o = 16; o; o >>= 1) var += __shfl_xor(var, o, 32);
    float inv = rsqrtf(var * (1.f / (float)DMODEL) + 1e-5f);
#pragma unroll
    for (int i = 0; i < 16; ++i) {
        int c = lane * 16 + i;
        out[(size_t)row * DMODEL + c] = (v[i] - mu) * inv * g[c] + be[c];
    }
}

// ---------------------------------------------------------------------------
// Output projection (N=21, last PREDLEN positions) — tiny, scalar fp32
// ---------------------------------------------------------------------------
__launch_bounds__(256)
__global__ void proj_kernel(const float* __restrict__ xd, const float* __restrict__ w,
                            const float* __restrict__ bias, float* __restrict__ out) {
    int idx = blockIdx.x * 256 + threadIdx.x;
    if (idx >= BATCH * PREDLEN * COUT) return;
    int c = idx % COUT;
    int t = (idx / COUT) % PREDLEN;
    int b = idx / (COUT * PREDLEN);
    const float* xr = xd + ((size_t)b * SEQ + (SEQ - PREDLEN) + t) * DMODEL;
    float acc = bias[c];
    for (int k = 0; k < DMODEL; ++k) acc += xr[k] * w[k * COUT + c];
    out[idx] = acc;
}

// ---------------------------------------------------------------------------
// Host orchestration
// ---------------------------------------------------------------------------
extern "C" void kernel_launch(void* const* d_in, const int* in_sizes, int n_in,
                              void* d_out, int out_size, void* d_ws, size_t ws_size,
                              hipStream_t stream) {
    (void)in_sizes; (void)n_in; (void)out_size; (void)ws_size;
    hipFuncSetAttribute((const void*)attn_kernel,
                        hipFuncAttributeMaxDynamicSharedMemorySize, ATTN_SMEM);

    auto F = [&](int i) { return (const float*)d_in[i]; };
    char* ws = (char*)d_ws;

    // ---- workspace layout ----
    size_t off = 0;
    auto take = [&](size_t bytes) { size_t r = off; off += (bytes + 255) & ~(size_t)255; return r; };
    const size_t SZ_DD = (size_t)DMODEL * DMODEL * 2;     // bf16 512x512
    const size_t SZ_DF = (size_t)DMODEL * DFFN * 2;       // bf16 512x2048
    const size_t SZ_ACT = (size_t)MROWS * DMODEL * 4;     // 32 MB
    size_t pEnc[2][6];
    for (int l = 0; l < 2; ++l) {
        for (int j = 0; j < 4; ++j) pEnc[l][j] = take(SZ_DD);
        pEnc[l][4] = take(SZ_DF);
        pEnc[l][5] = take(SZ_DF);
    }
    size_t pSelf[4], pCross[4];
    for (int j = 0; j < 4; ++j) pSelf[j] = take(SZ_DD);
    for (int j = 0; j < 4; ++j) pCross[j] = take(SZ_DD);
    size_t pDW1 = take(SZ_DF), pDW2 = take(SZ_DF);
    size_t oME = take((size_t)BATCH * SEQ * 4);
    size_t oMD = take((size_t)BATCH * SEQ * 4);
    size_t oXE = take(SZ_ACT), oXD = take(SZ_ACT), oEO = take(SZ_ACT);
    size_t oQ = take(SZ_ACT), oK = take(SZ_ACT), oV = take(SZ_ACT), oT = take(SZ_ACT);
    size_t oMID = take((size_t)4096 * DFFN * 4);          // FFN chunk scratch (32 MB)

    float* XE = (float*)(ws + oXE); float* XD = (float*)(ws + oXD);
    float* EO = (float*)(ws + oEO);
    float* Qb = (float*)(ws + oQ); float* Kb = (float*)(ws + oK);
    float* Vb = (float*)(ws + oV); float* Tb = (float*)(ws + oT);
    float* MID = (float*)(ws + oMID);
    float* ME = (float*)(ws + oME); float* MD = (float*)(ws + oMD);

    // ---- helpers ----
    auto pack = [&](int idx, size_t dst, int K, int N) {
        pack_wt_kernel<<<(K * N + 255) / 256, 256, 0, stream>>>(F(idx), (__bf16*)(ws + dst), K, N);
    };
    auto gemm = [&](const float* A, size_t wt, int bidx, float* C, int M, int N, int K, int gelu) {
        gemm_bf16_kernel<<<dim3(M / 128, N / 128), 256, 0, stream>>>(
            A, (const __bf16*)(ws + wt), F(bidx), C, M, N, K, gelu);
    };
    auto attn = [&](const float* q, const float* k, const float* v, const float* bias,
                    float* o, int causal) {
        attn_kernel<<<dim3(SEQ / 64, NHEAD, BATCH), 128, ATTN_SMEM, stream>>>(q, k, v, bias, o, causal);
    };
    auto addln = [&](const float* x, const float* y, int gi, int bi, float* o) {
        add_ln_kernel<<<MROWS / 8, 256, 0, stream>>>(x, y, F(gi), F(bi), o, MROWS);
    };
    auto ffn = [&](float* X, size_t w1, int b1, size_t w2, int b2) {
        for (int c = 0; c < 4; ++c) {
            gemm(X + (size_t)c * 4096 * DMODEL, w1, b1, MID, 4096, DFFN, DMODEL, 1);
            gemm(MID, w2, b2, Tb + (size_t)c * 4096 * DMODEL, 4096, DMODEL, DFFN, 0);
        }
    };

    // ---- input index map (setup_inputs insertion order) ----
    // 0 x_enc, 1 x_mark_enc, 2 x_dec, 3 x_mark_dec,
    // 4/5 enc_emb conv/temp, 6/7 dec_emb conv/temp,
    // enc layer l base=8+16*l: Wq,bq,Wk,bk,Wv,bv,Wo,bo,ffw1,ffb1,ffw2,ffb2,n1g,n1b,n2g,n2b
    // dec layer base=40: self(8), cross(8), ffw1..ffb2(4), n1g..n3b(6)
    // 66/67 enc_norm, 68/69 dec_norm, 70/71 proj

    // ---- pack all GEMM weights (fp32 -> transposed bf16) ----
    for (int l = 0; l < 2; ++l) {
        int base = 8 + 16 * l;
        for (int j = 0; j < 4; ++j) pack(base + 2 * j, pEnc[l][j], DMODEL, DMODEL);
        pack(base + 8, pEnc[l][4], DMODEL, DFFN);
        pack(base + 10, pEnc[l][5], DFFN, DMODEL);
    }
    for (int j = 0; j < 4; ++j) pack(40 + 2 * j, pSelf[j], DMODEL, DMODEL);
    for (int j = 0; j < 4; ++j) pack(48 + 2 * j, pCross[j], DMODEL, DMODEL);
    pack(56, pDW1, DMODEL, DFFN);
    pack(58, pDW2, DFFN, DMODEL);

    // ---- momentum biases + embeddings ----
    momentum_kernel<<<(BATCH * SEQ + 255) / 256, 256, 0, stream>>>(F(0), ME);
    momentum_kernel<<<(BATCH * SEQ + 255) / 256, 256, 0, stream>>>(F(2), MD);
    embed_kernel<<<(BATCH * SEQ * DMODEL + 255) / 256, 256, 0, stream>>>(F(0), F(1), F(4), F(5), XE);
    embed_kernel<<<(BATCH * SEQ * DMODEL + 255) / 256, 256, 0, stream>>>(F(2), F(3), F(6), F(7), XD);

    // ---- encoder ----
    for (int l = 0; l < 2; ++l) {
        int base = 8 + 16 * l;
        gemm(XE, pEnc[l][0], base + 1, Qb, MROWS, DMODEL, DMODEL, 0);
        gemm(XE, pEnc[l][1], base + 3, Kb, MROWS, DMODEL, DMODEL, 0);
        gemm(XE, pEnc[l][2], base + 5, Vb, MROWS, DMODEL, DMODEL, 0);
        attn(Qb, Kb, Vb, ME, Tb, 0);
        gemm(Tb, pEnc[l][3], base + 7, Qb, MROWS, DMODEL, DMODEL, 0);
        addln(XE, Qb, base + 12, base + 13, XE);
        ffn(XE, pEnc[l][4], base + 9, pEnc[l][5], base + 11);
        addln(XE, Tb, base + 14, base + 15, XE);
    }
    addln(XE, nullptr, 66, 67, EO);   // enc_norm -> enc_out

    // ---- decoder ----
    {
        gemm(XD, pSelf[0], 41, Qb, MROWS, DMODEL, DMODEL, 0);
        gemm(XD, pSelf[1], 43, Kb, MROWS, DMODEL, DMODEL, 0);
        gemm(XD, pSelf[2], 45, Vb, MROWS, DMODEL, DMODEL, 0);
        attn(Qb, Kb, Vb, MD, Tb, 1);                       // causal self-attn
        gemm(Tb, pSelf[3], 47, Qb, MROWS, DMODEL, DMODEL, 0);
        addln(XD, Qb, 60, 61, XD);

        gemm(XD, pCross[0], 49, Qb, MROWS, DMODEL, DMODEL, 0);
        gemm(EO, pCross[1], 51, Kb, MROWS, DMODEL, DMODEL, 0);
        gemm(EO, pCross[2], 53, Vb, MROWS, DMODEL, DMODEL, 0);
        attn(Qb, Kb, Vb, ME, Tb, 0);                       // cross-attn
        gemm(Tb, pCross[3], 55, Qb, MROWS, DMODEL, DMODEL, 0);
        addln(XD, Qb, 62, 63, XD);

        ffn(XD, pDW1, 57, pDW2, 59);
        addln(XD, Tb, 64, 65, XD);
    }
    addln(XD, nullptr, 68, 69, XD);   // dec_norm (in place, row-wise safe)

    // ---- projection ----
    proj_kernel<<<(BATCH * PREDLEN * COUT + 255) / 256, 256, 0, stream>>>(
        XD, F(70), F(71), (float*)d_out);
}
